// AKT_6571299963214
// MI455X (gfx1250) — matrix operand order repaired and checked by reading, not compiled, not run on hardware
//
#include <hip/hip_runtime.h>
#include <math.h>
#include <stdint.h>

// ---------------------------------------------------------------------------
// AKT forward for MI455X (gfx1250, wave32, WMMA, TDM).
// GEMMs: v_wmma_f32_16x16x32_bf16, 128x128x64 tiles. Weights pre-transposed
// to [N,K] so A- and B-tiles stage identically. Tiles staged by
// TENSOR_LOAD_TO_LDS with pad_enable producing the padded LDS stride
// directly; DOUBLE-BUFFERED: tile i+1's TDM is in flight (TENSORcnt=2..4)
// while tile i is computed (s_wait_tensorcnt 2). Fallback = batched b128
// loads, single buffer. Attention fused per (b,h,16 rows): scores in LDS,
// decay + double softmax, P@V via WMMA. Workspace: ~126 MB.
// ---------------------------------------------------------------------------

typedef __attribute__((ext_vector_type(16))) __bf16 bf16x16;
typedef __attribute__((ext_vector_type(8)))  float  f32x8;

union Frag { bf16x16 v; uint4 q[2]; unsigned short u[16]; };

constexpr int BB  = 16;    // batch
constexpr int SS  = 512;   // seq
constexpr int DDm = 512;   // model dim
constexpr int HH  = 8;     // heads
constexpr int DKK = 64;    // head dim
constexpr int FFd = 2048;  // ffn dim

__device__ __forceinline__ unsigned short f2bf(float f) {
  unsigned int u = __float_as_uint(f);
  u += 0x7FFFu + ((u >> 16) & 1u);           // round-to-nearest-even
  return (unsigned short)(u >> 16);
}

#if defined(__has_builtin)
#if __has_builtin(__builtin_amdgcn_tensor_load_to_lds) && \
    __has_builtin(__builtin_amdgcn_s_wait_tensorcnt)
#define USE_TDM 1
#endif
#endif

#ifdef USE_TDM
typedef __attribute__((ext_vector_type(4))) unsigned int u32x4;
typedef __attribute__((ext_vector_type(8))) int          i32x8;
typedef __attribute__((ext_vector_type(4))) int          i32x4;

// Issue one TDM 2D tile load: 128 rows x 64 bf16, row stride `ld` elements in
// global, padded to 72-half rows in LDS (pad_interval=32 DW, pad_amount=4 DW).
// D# layout per CDNA5 ISA.
__device__ __forceinline__ void tdm_load_tile(unsigned lds_off,
                                              const unsigned short* gbase,
                                              int row0, int col0, int ld) {
  unsigned long long ga =
      (unsigned long long)(uintptr_t)gbase +
      ((unsigned long long)row0 * (unsigned)ld + (unsigned)col0) * 2ull;
  u32x4 g0;
  g0[0] = 1u;                                   // count=1, user descriptor
  g0[1] = lds_off;                              // lds_addr (bytes)
  g0[2] = (unsigned)(ga & 0xFFFFFFFFull);       // global_addr[31:0]
  g0[3] = (unsigned)((ga >> 32) & 0x1FFFFFFull) // global_addr[56:32]
          | (2u << 30);                         // type = 2 ("image")
  i32x8 g1;
  g1[0] = (int)((1u << 16)      // data_size = 2 bytes
                | (1u << 20)    // pad_enable
                | (4u << 22)    // pad_interval: 32 DWORDs (=128B of data)
                | (3u << 25));  // pad_amount: 4 DWORDs (=16B pad) -> 144B rows
  const unsigned td0 = 1u << 20, td1 = 1u << 20;  // huge dims: no OOB clipping
  g1[1] = (int)((td0 & 0xFFFFu) << 16);                        // dim0 lo16
  g1[2] = (int)(((td0 >> 16) & 0xFFFFu) | ((td1 & 0xFFFFu) << 16));
  g1[3] = (int)(((td1 >> 16) & 0xFFFFu) | (64u << 16));        // tile_dim0=64
  g1[4] = 128;                                  // tile_dim1=128, tile_dim2=0
  g1[5] = ld;                                   // tensor_dim0_stride lo32
  g1[6] = 0;
  g1[7] = 0;
  i32x4 z4 = {0, 0, 0, 0};
#if __clang_major__ >= 23
  i32x8 z8 = {0, 0, 0, 0, 0, 0, 0, 0};
  __builtin_amdgcn_tensor_load_to_lds(g0, g1, z4, z4, z8, 0);
#else
  __builtin_amdgcn_tensor_load_to_lds(g0, g1, z4, z4, 0);
#endif
}
#define NBUF 2
#else
#define NBUF 1
#endif  // USE_TDM

// ---------------------------------------------------------------------------
// f32 -> bf16 convert (linear)
// ---------------------------------------------------------------------------
__global__ void cvt_kernel(const float* __restrict__ in,
                           unsigned short* __restrict__ out, size_t n) {
  size_t i = (size_t)blockIdx.x * blockDim.x + threadIdx.x;
  size_t stride = (size_t)gridDim.x * blockDim.x;
  for (; i < n; i += stride) out[i] = f2bf(in[i]);
}

// ---------------------------------------------------------------------------
// f32 [L,K,N] -> bf16 [L,N,K] transpose (32x32 LDS tiles, coalesced both ways)
// ---------------------------------------------------------------------------
__global__ __launch_bounds__(256) void cvtT_kernel(
    const float* __restrict__ in, unsigned short* __restrict__ out,
    int K, int N) {
  __shared__ float t[32][33];
  const int l = blockIdx.z;
  const int k0 = blockIdx.y * 32, n0 = blockIdx.x * 32;
  const int tx = threadIdx.x & 31, ty = threadIdx.x >> 5;  // 32 x 8
  const float* src = in + (size_t)l * K * N;
  unsigned short* dst = out + (size_t)l * K * N;
#pragma unroll
  for (int r = ty; r < 32; r += 8)
    t[r][tx] = src[(size_t)(k0 + r) * N + n0 + tx];
  __syncthreads();
#pragma unroll
  for (int r = ty; r < 32; r += 8)
    dst[(size_t)(n0 + r) * K + k0 + tx] = f2bf(t[tx][r]);
}

// ---------------------------------------------------------------------------
// Tiled bf16 GEMM: C[M,N] = act(A[M,K] @ Bt[N,K]^T + bias [+ resid])
// ---------------------------------------------------------------------------
#define BM 128
#define BN 128
#define BKK 64
#define LDT 72   // LDS row stride in halfs: 128B data + 16B pad (TDM-matched)

__global__ __launch_bounds__(256) void gemm_bf16(
    const unsigned short* __restrict__ A, const unsigned short* __restrict__ Bt,
    const float* __restrict__ bias, const float* __restrict__ resid,
    float* __restrict__ outF, unsigned short* __restrict__ outH,
    int M, int N, int K, int relu, int head_mode) {
  __shared__ unsigned short lA[NBUF][BM * LDT];   // [buf][row][k]
  __shared__ unsigned short lB[NBUF][BN * LDT];   // [buf][col][k]

  const int tid  = threadIdx.x;
  const int lane = tid & 31;
  const int wv   = tid >> 5;
  const int wM   = (wv & 1) * 64;
  const int wN   = (wv >> 1) * 32;
  const int m0   = blockIdx.y * BM;
  const int n0   = blockIdx.x * BN;
  const int r16  = lane & 15;
  const int ab   = (lane < 16) ? 0 : 8;    // A frag K sub-base
  const int bb   = (lane < 16) ? 0 : 8;    // B frag K sub-base

  f32x8 z = {0.f, 0.f, 0.f, 0.f, 0.f, 0.f, 0.f, 0.f};
  f32x8 acc[4][2];
  for (int mi = 0; mi < 4; ++mi)
    for (int ni = 0; ni < 2; ++ni) acc[mi][ni] = z;

  const int nIter = K / BKK;

#ifdef USE_TDM
  const unsigned lAo[2] = {(unsigned)(uintptr_t)&lA[0][0],
                           (unsigned)(uintptr_t)&lA[NBUF - 1][0]};
  const unsigned lBo[2] = {(unsigned)(uintptr_t)&lB[0][0],
                           (unsigned)(uintptr_t)&lB[NBUF - 1][0]};
  if (wv == 0) {  // prologue: tile 0 into buffer 0
    tdm_load_tile(lAo[0], A,  m0, 0, K);
    tdm_load_tile(lBo[0], Bt, n0, 0, K);
  }
#endif

  for (int itK = 0; itK < nIter; ++itK) {
    const int k0 = itK * BKK;
#ifdef USE_TDM
    const int cur = itK & 1;
    if (wv == 0) {
      if (itK + 1 < nIter) {  // keep next tile's DMA in flight
        tdm_load_tile(lAo[cur ^ 1], A,  m0, k0 + BKK, K);
        tdm_load_tile(lBo[cur ^ 1], Bt, n0, k0 + BKK, K);
        __builtin_amdgcn_s_wait_tensorcnt(2);   // tile itK complete
      } else {
        __builtin_amdgcn_s_wait_tensorcnt(0);
      }
    }
    if (k0 + 2 * BKK < K) {  // warm GL2 two tiles ahead
      int pr = m0 + (tid >> 1); if (pr >= M) pr = M - 1;
      __builtin_prefetch(A + (size_t)pr * K + k0 + 2 * BKK, 0, 0);
      int pn = n0 + (tid >> 1); if (pn >= N) pn = N - 1;
      __builtin_prefetch(Bt + (size_t)pn * K + k0 + 2 * BKK, 0, 0);
    }
    __syncthreads();
#else
    const int cur = 0;
    // batched staging: issue all 8 b128 loads, one wait, then 8 LDS stores
    uint4 ra[4], rb[4];
#pragma unroll
    for (int c = 0; c < 4; ++c) {
      int lin = c * 256 + tid;
      int row = lin >> 3;
      int cb  = (lin & 7) << 3;
      ra[c] = *(const uint4*)(A  + (size_t)(m0 + row) * K + k0 + cb);
      rb[c] = *(const uint4*)(Bt + (size_t)(n0 + row) * K + k0 + cb);
    }
    if (k0 + BKK < K) {
      int pr = m0 + (tid >> 1); if (pr >= M) pr = M - 1;
      __builtin_prefetch(A + (size_t)pr * K + k0 + BKK, 0, 0);
      int pn = n0 + (tid >> 1); if (pn >= N) pn = N - 1;
      __builtin_prefetch(Bt + (size_t)pn * K + k0 + BKK, 0, 0);
    }
#pragma unroll
    for (int c = 0; c < 4; ++c) {
      int lin = c * 256 + tid;
      int row = lin >> 3;
      int cb  = (lin & 7) << 3;
      *(uint4*)(&lA[0][0] + row * LDT + cb) = ra[c];
      *(uint4*)(&lB[0][0] + row * LDT + cb) = rb[c];
    }
    __syncthreads();
#endif

    const unsigned short* curA = &lA[cur][0];
    const unsigned short* curB = &lB[cur][0];
#pragma unroll
    for (int kk = 0; kk < BKK; kk += 32) {
      Frag fa[4], fb[2];
#pragma unroll
      for (int mi = 0; mi < 4; ++mi) {
        const unsigned short* p = curA + (wM + mi * 16 + r16) * LDT + kk + ab;
        fa[mi].q[0] = *(const uint4*)p;         // K = kk+ab   .. +7
        fa[mi].q[1] = *(const uint4*)(p + 16);  // K = kk+16+ab.. +7
      }
#pragma unroll
      for (int ni = 0; ni < 2; ++ni) {
        const unsigned short* p = curB + (wN + ni * 16 + r16) * LDT + kk + bb;
        fb[ni].q[0] = *(const uint4*)p;         // K = kk+bb   .. +7
        fb[ni].q[1] = *(const uint4*)(p + 16);  // K = kk+16+bb.. +7
      }
#pragma unroll
      for (int mi = 0; mi < 4; ++mi)
#pragma unroll
        for (int ni = 0; ni < 2; ++ni)
          acc[mi][ni] = __builtin_amdgcn_wmma_f32_16x16x32_bf16(
              false, fa[mi].v, false, fb[ni].v, (short)0, acc[mi][ni],
              false, false);
    }
    __syncthreads();  // all waves done reading buf[cur] before its reuse
  }

  // epilogue: bias + residual + relu, f32 and/or bf16 (optionally head split)
#pragma unroll
  for (int mi = 0; mi < 4; ++mi)
    for (int ni = 0; ni < 2; ++ni)
      for (int r = 0; r < 8; ++r) {
        int grow = m0 + wM + mi * 16 + r + ((lane < 16) ? 0 : 8);
        int gcol = n0 + wN + ni * 16 + r16;
        float val = acc[mi][ni][r] + bias[gcol];
        if (resid) val += resid[(size_t)grow * N + gcol];
        if (relu)  val = fmaxf(val, 0.f);
        if (head_mode) {
          int bbk = grow / SS, s = grow % SS;
          int hh  = gcol / DKK, dk = gcol % DKK;
          outH[(((size_t)bbk * HH + hh) * SS + s) * DKK + dk] = f2bf(val);
        } else {
          size_t idx = (size_t)grow * N + gcol;
          if (outF) outF[idx] = val;
          if (outH) outH[idx] = f2bf(val);
        }
      }
}

// ---------------------------------------------------------------------------
// Fused decay-attention: one wave per (b, h, 16 query rows).
// K == Q (kq_same). scores kept in LDS; causal tiles only.
// mask_excl: 0 -> j<=i valid, 1 -> j<i valid.
// Output written merged-head bf16 [B*S, D].
// ---------------------------------------------------------------------------
__global__ __launch_bounds__(32) void attn_kernel(
    const unsigned short* __restrict__ qh, const unsigned short* __restrict__ vh,
    const float* __restrict__ gam, unsigned short* __restrict__ out,
    int mask_excl) {
  __shared__ float          sc[16][520];  // scores (padded f32)
  __shared__ unsigned short pa[16][528];  // probabilities (padded bf16)

  const int lane = threadIdx.x;
  const int it = blockIdx.x, h = blockIdx.y, b = blockIdx.z;
  const int i0 = it * 16;
  const unsigned short* q = qh + ((size_t)(b * HH + h)) * SS * DKK;
  const unsigned short* v = vh + ((size_t)(b * HH + h)) * SS * DKK;
  const float g = -log1pf(__expf(gam[h]));  // -softplus(gamma)

  const int r16 = lane & 15;
  const int ab  = (lane < 16) ? 0 : 16;
  const int bb  = (lane < 16) ? 0 : 16;

  // Q fragments for K-dim 0..63 (two 16x32 A-frags)
  Frag aq0, aq1;
#pragma unroll
  for (int i = 0; i < 8; ++i) {
    const unsigned short* qr = q + (size_t)(i0 + r16) * DKK;
    aq0.u[i]     = qr[ab + i];
    aq0.u[8 + i] = qr[8 + ab + i];
    aq1.u[i]     = qr[32 + ab + i];
    aq1.u[8 + i] = qr[40 + ab + i];
  }

  const float scale = 0.125f;  // 1/sqrt(64)
  // ---- S = Q K^T over causal tiles, masked into LDS ----
  for (int jt = 0; jt <= it; ++jt) {
    Frag b0, b1;
    int j = jt * 16 + r16;
#pragma unroll
    for (int i = 0; i < 16; ++i) {
      const unsigned short* kr = q + (size_t)j * DKK;   // K == Q
      b0.u[i] = kr[bb + i];
      b1.u[i] = kr[32 + bb + i];
    }
    f32x8 acc = {0.f, 0.f, 0.f, 0.f, 0.f, 0.f, 0.f, 0.f};
    acc = __builtin_amdgcn_wmma_f32_16x16x32_bf16(false, aq0.v, false, b0.v,
                                                  (short)0, acc, false, false);
    acc = __builtin_amdgcn_wmma_f32_16x16x32_bf16(false, aq1.v, false, b1.v,
                                                  (short)0, acc, false, false);
#pragma unroll
    for (int r = 0; r < 8; ++r) {
      int row = r + ((lane < 16) ? 0 : 8);
      int gi = i0 + row;
      bool valid = mask_excl ? (j < gi) : (j <= gi);
      sc[row][j] = valid ? acc[r] * scale : -1e30f;
    }
  }
  __syncthreads();

  const int ncols = (it + 1) * 16;
  const int nK = (ncols + 31) & ~31;

  // ---- per-row: softmax -> cumsum decay -> rescore -> softmax -> bf16 ----
  if (lane < 16) {
    int row = lane, gi = i0 + row;
    float m = -3e38f;
    for (int j = 0; j < ncols; ++j) m = fmaxf(m, sc[row][j]);
    if (m < -1e29f) {  // fully masked row: uniform fallback
      unsigned short ub = f2bf(1.0f / (float)ncols);
      for (int j = 0; j < nK; ++j) pa[row][j] = (j < ncols) ? ub : 0;
    } else {
      float s1 = 0.f;
      for (int j = 0; j < ncols; ++j) s1 += __expf(sc[row][j] - m);
      float inv = 1.f / s1, cum = 0.f, m2 = -3e38f;
      for (int j = 0; j < ncols; ++j) {
        float sv = sc[row][j];
        float p = __expf(sv - m) * inv;      // masked -> ~0
        cum += p;
        float rem = fmaxf(1.f - cum, 0.f);   // disttot - distcum
        float pos = fabsf((float)(j - gi));
        float eff = __expf(sqrtf(rem * pos) * g);
        eff = fminf(fmaxf(eff, 1e-5f), 1e5f);
        float ns = (sv > -1e29f) ? sv * eff : -1e30f;
        sc[row][j] = ns;
        m2 = fmaxf(m2, ns);
      }
      float s2 = 0.f;
      for (int j = 0; j < ncols; ++j) s2 += __expf(sc[row][j] - m2);
      float inv2 = 1.f / s2;
      for (int j = 0; j < nK; ++j)
        pa[row][j] = (j < ncols) ? f2bf(__expf(sc[row][j] - m2) * inv2) : 0;
    }
  }
  __syncthreads();

  // ---- O = P V via WMMA (A from LDS probs, B strided from V) ----
  f32x8 accv[4];
  f32x8 z = {0.f, 0.f, 0.f, 0.f, 0.f, 0.f, 0.f, 0.f};
  for (int ni = 0; ni < 4; ++ni) accv[ni] = z;

  for (int jt2 = 0; jt2 * 32 < nK; ++jt2) {
    Frag fa;
#pragma unroll
    for (int i = 0; i < 8; ++i) {
      fa.u[i]     = pa[r16][jt2 * 32 + ab + i];
      fa.u[8 + i] = pa[r16][jt2 * 32 + 8 + ab + i];
    }
#pragma unroll
    for (int ni = 0; ni < 4; ++ni) {
      Frag fb;
      int dk = ni * 16 + r16;
#pragma unroll
      for (int i = 0; i < 16; ++i)
        fb.u[i] = v[(size_t)(jt2 * 32 + bb + i) * DKK + dk];
      accv[ni] = __builtin_amdgcn_wmma_f32_16x16x32_bf16(
          false, fa.v, false, fb.v, (short)0, accv[ni], false, false);
    }
  }

#pragma unroll
  for (int ni = 0; ni < 4; ++ni)
    for (int r = 0; r < 8; ++r) {
      int gi = i0 + r + ((lane < 16) ? 0 : 8);
      int dk = ni * 16 + r16;
      out[((size_t)b * SS + gi) * DDm + h * DKK + dk] = f2bf(accv[ni][r]);
    }
}

// ---------------------------------------------------------------------------
// LayerNorm over D=512, one 128-thread block per row. In-place safe.
// ---------------------------------------------------------------------------
__global__ __launch_bounds__(128) void ln_kernel(
    const float* __restrict__ in, const float* __restrict__ gam,
    const float* __restrict__ bet, float* __restrict__ outF,
    unsigned short* __restrict__ outH) {
  __shared__ float red[128];
  const int row = blockIdx.x, tid = threadIdx.x;
  const float* p = in + (size_t)row * DDm;
  float4 x = *(const float4*)(p + tid * 4);

  red[tid] = x.x + x.y + x.z + x.w;
  __syncthreads();
  for (int o = 64; o > 0; o >>= 1) {
    if (tid < o) red[tid] += red[tid + o];
    __syncthreads();
  }
  float mean = red[0] * (1.f / 512.f);
  __syncthreads();

  float d0 = x.x - mean, d1 = x.y - mean, d2 = x.z - mean, d3 = x.w - mean;
  red[tid] = d0 * d0 + d1 * d1 + d2 * d2 + d3 * d3;
  __syncthreads();
  for (int o = 64; o > 0; o >>= 1) {
    if (tid < o) red[tid] += red[tid + o];
    __syncthreads();
  }
  float rs = rsqrtf(red[0] * (1.f / 512.f) + 1e-5f);

  float dv[4] = {d0, d1, d2, d3};
#pragma unroll
  for (int i = 0; i < 4; ++i) {
    int c = tid * 4 + i;
    float y = dv[i] * rs * gam[c] + bet[c];
    outF[(size_t)row * DDm + c] = y;
    if (outH) outH[(size_t)row * DDm + c] = f2bf(y);
  }
}

// ---------------------------------------------------------------------------
// Host orchestration
// ---------------------------------------------------------------------------
extern "C" void kernel_launch(void* const* d_in, const int* in_sizes, int n_in,
                              void* d_out, int out_size, void* d_ws,
                              size_t ws_size, hipStream_t stream) {
  (void)in_sizes; (void)n_in; (void)out_size; (void)ws_size;
  const float* q_embed  = (const float*)d_in[0];
  const float* qa_embed = (const float*)d_in[1];
  const float* Wk = (const float*)d_in[2];
  const float* bk = (const float*)d_in[3];
  const float* Wv = (const float*)d_in[4];
  const float* bv = (const float*)d_in[5];
  const float* Wo = (const float*)d_in[6];
  const float* bo = (const float*)d_in[7];
  const float* gammas = (const float*)d_in[8];
  const float* ln1_g = (const float*)d_in[9];
  const float* ln1_b = (const float*)d_in[10];
  const float* W1 = (const float*)d_in[11];
  const float* b1 = (const float*)d_in[12];
  const float* W2 = (const float*)d_in[13];
  const float* b2 = (const float*)d_in[14];
  const float* ln2_g = (const float*)d_in[15];
  const float* ln2_b = (const float*)d_in[16];

  const size_t BSD = (size_t)BB * SS * DDm;     // 4,194,304
  const size_t BSF = (size_t)BB * SS * FFd;     // 16,777,216
  const size_t WDD = (size_t)3 * DDm * DDm;
  const size_t WDF = (size_t)3 * DDm * FFd;
  const int    M   = BB * SS;                   // 8192

  char* ws = (char*)d_ws;
  size_t off = 0;
  auto alloc = [&](size_t bytes) {
    size_t o = off;
    off += (bytes + 255) & ~(size_t)255;
    return o;
  };
  float* y_f = (float*)(ws + alloc(BSD * 4));
  float* x_f = (float*)(ws + alloc(BSD * 4));
  unsigned short* y_h    = (unsigned short*)(ws + alloc(BSD * 2));
  unsigned short* x_h    = (unsigned short*)(ws + alloc(BSD * 2));
  unsigned short* q_h    = (unsigned short*)(ws + alloc(BSD * 2));
  unsigned short* v_h    = (unsigned short*)(ws + alloc(BSD * 2));
  unsigned short* attn_h = (unsigned short*)(ws + alloc(BSD * 2));
  unsigned short* hid_h  = (unsigned short*)(ws + alloc(BSF * 2));
  unsigned short* wkT = (unsigned short*)(ws + alloc(WDD * 2));  // [L,N=D,K=D]
  unsigned short* wvT = (unsigned short*)(ws + alloc(WDD * 2));
  unsigned short* woT = (unsigned short*)(ws + alloc(WDD * 2));
  unsigned short* w1T = (unsigned short*)(ws + alloc(WDF * 2));  // [L,N=F,K=D]
  unsigned short* w2T = (unsigned short*)(ws + alloc(WDF * 2));  // [L,N=D,K=F]

  auto cvt = [&](const float* in, unsigned short* out, size_t n) {
    int blocks = (int)((n + 1023) / 1024);
    cvt_kernel<<<blocks, 1024, 0, stream>>>(in, out, n);
  };
  auto cvtT = [&](const float* in, unsigned short* out, int K, int N) {
    dim3 g(N / 32, K / 32, 3);
    cvtT_kernel<<<g, 256, 0, stream>>>(in, out, K, N);
  };
  auto gemm = [&](const unsigned short* A, const unsigned short* Bt,
                  const float* bias, const float* resid, float* oF,
                  unsigned short* oH, int Mm, int N, int K, int relu,
                  int head) {
    dim3 g(N / BN, Mm / BM);
    gemm_bf16<<<g, 256, 0, stream>>>(A, Bt, bias, resid, oF, oH, Mm, N, K,
                                     relu, head);
  };
  auto attn = [&](const unsigned short* qp, const unsigned short* vp,
                  const float* gm, unsigned short* op, int excl) {
    attn_kernel<<<dim3(SS / 16, HH, BB), 32, 0, stream>>>(qp, vp, gm, op, excl);
  };
  auto ln = [&](const float* in, const float* g, const float* b, float* oF,
                unsigned short* oH) {
    ln_kernel<<<M, 128, 0, stream>>>(in, g, b, oF, oH);
  };

  // weight conversion + transpose to [N,K] bf16; embeddings to bf16
  cvtT(Wk, wkT, DDm, DDm);
  cvtT(Wv, wvT, DDm, DDm);
  cvtT(Wo, woT, DDm, DDm);
  cvtT(W1, w1T, DDm, FFd);   // [D,F] -> [F,D]
  cvtT(W2, w2T, FFd, DDm);   // [F,D] -> [D,F]
  cvt(qa_embed, y_h, BSD);
  cvt(q_embed,  x_h, BSD);

  const size_t DD2 = (size_t)DDm * DDm, DF = (size_t)DDm * FFd;

  // ---- block 0: knowledge encoder (y = y, mask1, FFN) ----
  gemm(y_h, wkT, bk, nullptr, nullptr, q_h, M, DDm, DDm, 0, 1);
  gemm(y_h, wvT, bv, nullptr, nullptr, v_h, M, DDm, DDm, 0, 1);
  attn(q_h, v_h, gammas, attn_h, 0);
  gemm(attn_h, woT, bo, qa_embed, y_f, nullptr, M, DDm, DDm, 0, 0);
  ln(y_f, ln1_g, ln1_b, y_f, y_h);
  gemm(y_h, w1T, b1, nullptr, nullptr, hid_h, M, FFd, DDm, 1, 0);
  gemm(hid_h, w2T, b2, y_f, y_f, nullptr, M, DDm, FFd, 0, 0);
  ln(y_f, ln2_g, ln2_b, y_f, y_h);

  // ---- block 1: question encoder (x = x, mask1, no FFN) ----
  gemm(x_h, wkT + DD2, bk + DDm, nullptr, nullptr, q_h, M, DDm, DDm, 0, 1);
  gemm(x_h, wvT + DD2, bv + DDm, nullptr, nullptr, v_h, M, DDm, DDm, 0, 1);
  attn(q_h, v_h, gammas + HH, attn_h, 0);
  gemm(attn_h, woT + DD2, bo + DDm, q_embed, x_f, nullptr, M, DDm, DDm, 0, 0);
  ln(x_f, ln1_g + DDm, ln1_b + DDm, x_f, x_h);

  // ---- block 2: knowledge retriever (q=x, v=y, mask0, FFN) ----
  gemm(x_h, wkT + 2 * DD2, bk + 2 * DDm, nullptr, nullptr, q_h, M, DDm, DDm, 0, 1);
  gemm(y_h, wvT + 2 * DD2, bv + 2 * DDm, nullptr, nullptr, v_h, M, DDm, DDm, 0, 1);
  attn(q_h, v_h, gammas + 2 * HH, attn_h, 1);
  gemm(attn_h, woT + 2 * DD2, bo + 2 * DDm, x_f, x_f, nullptr, M, DDm, DDm, 0, 0);
  ln(x_f, ln1_g + 2 * DDm, ln1_b + 2 * DDm, x_f, x_h);
  gemm(x_h, w1T + 2 * DF, b1 + 2 * FFd, nullptr, nullptr, hid_h, M, FFd, DDm, 1, 0);
  gemm(hid_h, w2T + 2 * DF, b2 + 2 * DDm, x_f, x_f, nullptr, M, DDm, FFd, 0, 0);
  ln(x_f, ln2_g + 2 * DDm, ln2_b + 2 * DDm, (float*)d_out, nullptr);
}
